// JITMetalmLSTMBlock_3006477107706
// MI455X (gfx1250) — matrix operand order repaired and checked by reading, not compiled, not run on hardware
//
#include <hip/hip_runtime.h>

// ---------------- problem constants ----------------
#define Bq 4
#define Sq 2048
#define DMq 1024
#define Hq 16
#define Dq 64
#define O3 3072          // 3*H*D
#define NCHUNK (Sq / 64) // 32

typedef __attribute__((ext_vector_type(16))) __bf16 bf16x16;
typedef __attribute__((ext_vector_type(8))) float f32x8;

union BFrag {
  bf16x16 v;
  uint4 q[2];
  unsigned short u[16];
};

__device__ inline unsigned short f2bf(float f) {
  unsigned int u = __float_as_uint(f);
  unsigned int r = (u + 0x7FFFu + ((u >> 16) & 1u)) >> 16;
  return (unsigned short)r;
}
__device__ inline float bf2f(unsigned short s) {
  return __uint_as_float(((unsigned int)s) << 16);
}

__device__ inline f32x8 wmma_bf16(const BFrag& a, const BFrag& b, f32x8 c) {
  return __builtin_amdgcn_wmma_f32_16x16x32_bf16(false, a.v, false, b.v,
                                                 (short)0, c, false, false);
}

// A-operand 16x32 bf16 fragment: lane (hf,m): elems 0..15 = K[16hf..16hf+15]
// (row pointer `base` at k-offset 0)
__device__ inline void load_afrag(BFrag& f, const unsigned short* base, int hf) {
  f.q[0] = *(const uint4*)(base + 16 * hf);
  f.q[1] = *(const uint4*)(base + 16 * hf + 8);
}
// B-operand 32x16 bf16 fragment: lane (hf,n): elems 0..15 = K[16hf..16hf+15]
__device__ inline void load_bfrag(BFrag& f, const unsigned short* base, int hf) {
  f.q[0] = *(const uint4*)(base + 16 * hf);
  f.q[1] = *(const uint4*)(base + 16 * hf + 8);
}

// CDNA5 async global->LDS copy (ASYNCcnt), 32 bytes per thread
__device__ inline void stage_async32(unsigned ldsOff, const unsigned short* g) {
  asm volatile("global_load_async_to_lds_b128 %0, %1, off"
               :: "v"(ldsOff), "v"(g) : "memory");
  asm volatile("global_load_async_to_lds_b128 %0, %1, off"
               :: "v"(ldsOff + 16u), "v"(g + 8) : "memory");
}
// Wait until at most `keep4` (0 or 4) of this wave's async ops are in flight.
__device__ inline void wait_async_keep4() {
  asm volatile("s_wait_asynccnt 4" ::: "memory");
}
__device__ inline void wait_async0() {
  asm volatile("s_wait_asynccnt 0" ::: "memory");
}

__device__ inline float wave_sum(float v) {
  for (int off = 16; off > 0; off >>= 1) v += __shfl_xor(v, off);
  return v;
}

// =====================================================================
// Kernel 1: RMSNorm -> xn (bf16), plus 48 gate dots + softcap sigmoid
// grid = B*S blocks of 256
// =====================================================================
__global__ __launch_bounds__(256) void rmsnorm_gates_kernel(
    const float* __restrict__ x, const float* __restrict__ ln_w,
    const float* __restrict__ w_gate, unsigned short* __restrict__ xn_bf,
    float* __restrict__ gates) {
  __shared__ float xs[DMq];
  __shared__ float red[8];
  const int tok = blockIdx.x;
  const int tid = threadIdx.x;

  const float4 v = ((const float4*)(x + (size_t)tok * DMq))[tid];
  float ss = v.x * v.x + v.y * v.y + v.z * v.z + v.w * v.w;
  ss = wave_sum(ss);
  if ((tid & 31) == 0) red[tid >> 5] = ss;
  __syncthreads();
  float tot = red[0] + red[1] + red[2] + red[3] + red[4] + red[5] + red[6] + red[7];
  float rstd = rsqrtf(tot * (1.0f / DMq) + 1e-6f);

  const int i0 = tid * 4;
  float o0 = v.x * rstd * ln_w[i0 + 0];
  float o1 = v.y * rstd * ln_w[i0 + 1];
  float o2 = v.z * rstd * ln_w[i0 + 2];
  float o3 = v.w * rstd * ln_w[i0 + 3];
  xs[i0 + 0] = o0; xs[i0 + 1] = o1; xs[i0 + 2] = o2; xs[i0 + 3] = o3;
  unsigned short* xo = xn_bf + (size_t)tok * DMq + i0;
  xo[0] = f2bf(o0); xo[1] = f2bf(o1); xo[2] = f2bf(o2); xo[3] = f2bf(o3);
  __syncthreads();

  const int wv = tid >> 5, ln = tid & 31;
  for (int g = wv; g < 48; g += 8) {
    const float* wg = w_gate + (size_t)g * DMq;
    float s = 0.f;
    for (int d = ln; d < DMq; d += 32) s += xs[d] * wg[d];
    s = wave_sum(s);
    if (ln == 0) {
      float c = 15.0f * tanhf(s * (1.0f / 15.0f));
      gates[(size_t)tok * 48 + g] = 1.0f / (1.0f + __expf(-c));
    }
  }
}

// =====================================================================
// Kernel 2: fp32 -> bf16 conversion (weights)
// =====================================================================
__global__ void cvt_bf16_kernel(const float* __restrict__ src,
                                unsigned short* __restrict__ dst, int n) {
  int i = blockIdx.x * blockDim.x + threadIdx.x;
  if (i < n) dst[i] = f2bf(src[i]);
}

// =====================================================================
// Kernel 3: C[m][n] = sum_k A[m][k]*B[n][k], bf16 in, fp32 acc.
// EP==0: store bf16.  EP==1: y = acc + residual, store fp32.
// Block tile 128x128, 8 waves of 2x4 16x16 WMMA tiles, K=1024.
// 3-stage software pipeline: two k-slices in flight via
// GLOBAL_LOAD_ASYNC_TO_LDS_B128, partial waits (s_wait_asynccnt 4).
// Each thread issues exactly 4 async ops per stage; async loads retire
// in order, so asynccnt<=4 <=> oldest stage fully landed.
// =====================================================================
template <int EP>
__global__ __launch_bounds__(256) void gemm_bf16_nt(
    const unsigned short* __restrict__ A, const unsigned short* __restrict__ Bm,
    unsigned short* __restrict__ outBf, float* __restrict__ outF,
    const float* __restrict__ resid, int N) {
  __shared__ alignas(16) unsigned short Asl[3][128 * 40];
  __shared__ alignas(16) unsigned short Bsl[3][128 * 40];
  const int tid = threadIdx.x;
  const int lane = tid & 31, wave = tid >> 5;
  const int hf = lane >> 4, lm = lane & 15;
  const int wm = wave >> 1, wn = wave & 1;
  const int m0 = blockIdx.x * 128, n0 = blockIdx.y * 128;
  const int rowL = tid >> 1, segL = tid & 1;

  // per-thread source pointers (k advances in the loop)
  const unsigned short* gA = A + (size_t)(m0 + rowL) * 1024 + segL * 16;
  const unsigned short* gB = Bm + (size_t)(n0 + rowL) * 1024 + segL * 16;
  // per-thread LDS destinations (low 32 bits of generic ptr == LDS offset)
  const unsigned thrOff = (unsigned)((rowL * 40 + segL * 16) * 2);
  const unsigned aBase = (unsigned)(uintptr_t)(&Asl[0][0]) + thrOff;
  const unsigned bBase = (unsigned)(uintptr_t)(&Bsl[0][0]) + thrOff;
  const unsigned bufStride = 128 * 40 * 2;  // bytes per buffer

  const f32x8 zero = {0.f, 0.f, 0.f, 0.f, 0.f, 0.f, 0.f, 0.f};
  f32x8 acc[2][4];
  for (int i = 0; i < 2; i++)
    for (int j = 0; j < 4; j++) acc[i][j] = zero;

  // prologue: stage k-slices 0 and 1 into buffers 0 and 1
  stage_async32(aBase + 0 * bufStride, gA + 0);
  stage_async32(bBase + 0 * bufStride, gB + 0);
  stage_async32(aBase + 1 * bufStride, gA + 32);
  stage_async32(bBase + 1 * bufStride, gB + 32);

  const int NIT = 1024 / 32;  // 32
  for (int it = 0; it < NIT; ++it) {
    // oldest in-flight stage (= it) must be fully landed
    if (it + 1 < NIT) wait_async_keep4();
    else wait_async0();
    __syncthreads();

    const int cur = it % 3;
    const unsigned short* As = Asl[cur];
    const unsigned short* Bs = Bsl[cur];
    BFrag af[2], bfr[4];
#pragma unroll
    for (int mi = 0; mi < 2; mi++)
      load_afrag(af[mi], As + (wm * 32 + mi * 16 + lm) * 40, hf);
#pragma unroll
    for (int ni = 0; ni < 4; ni++)
      load_bfrag(bfr[ni], Bs + (wn * 64 + ni * 16 + lm) * 40, hf);
#pragma unroll
    for (int mi = 0; mi < 2; mi++)
#pragma unroll
      for (int ni = 0; ni < 4; ni++)
        acc[mi][ni] = wmma_bf16(af[mi], bfr[ni], acc[mi][ni]);

    // prefetch stage it+2 into buffer (it+2)%3 (previously read at it-1,
    // which every wave finished before this iteration's barrier)
    if (it + 2 < NIT) {
      const int nxt = (it + 2) % 3;
      const int kb = (it + 2) * 32;
      stage_async32(aBase + nxt * bufStride, gA + kb);
      stage_async32(bBase + nxt * bufStride, gB + kb);
    }
  }

#pragma unroll
  for (int mi = 0; mi < 2; mi++)
#pragma unroll
    for (int ni = 0; ni < 4; ni++) {
      int mg = m0 + wm * 32 + mi * 16 + 8 * hf;
      int ng = n0 + wn * 64 + ni * 16 + lm;
#pragma unroll
      for (int r = 0; r < 8; r++) {
        float vv = acc[mi][ni][r];
        size_t idx = (size_t)(mg + r) * N + ng;
        if (EP == 0) outBf[idx] = f2bf(vv);
        else outF[idx] = vv + resid[idx];
      }
    }
}

// =====================================================================
// Kernel 4: chunked-parallel mLSTM scan, one block per (b,h), 4 waves.
// Chunk C=64: P[t][s] = (q_t.k_s) e^{la_t-la_s} i_s (masked s<=t) via WMMA,
// O = P.V + (e^{la_t} q_t).H0 via WMMA, H' = e^{la63}H0 + K~^T V via WMMA.
// =====================================================================
__global__ __launch_bounds__(128) void mlstm_scan_kernel(
    const unsigned short* __restrict__ qkv, const float* __restrict__ gates,
    const float* __restrict__ hidden0, unsigned short* __restrict__ att,
    float* __restrict__ hfinal) {
  __shared__ alignas(16) float Hs[64][68];          // Hs[e][d]
  __shared__ float la[64], ii[64], oo[64];
  __shared__ alignas(16) unsigned short Qs[64][72]; // e^{la_t} q_t[d]
  __shared__ alignas(16) unsigned short Kr[64][72]; // k_s[d]
  __shared__ alignas(16) unsigned short Kt[64][72]; // Kt[d][s] = w_s k_s[d]
  __shared__ alignas(16) unsigned short Vt[64][72]; // Vt[e][s] = v_s[e]
  __shared__ alignas(16) unsigned short Pm[64][72]; // masked probs (bf16)

  const int tid = threadIdx.x;
  const int lane = tid & 31, wave = tid >> 5;       // 4 waves
  const int hf = lane >> 4, lm = lane & 15;
  const int bh = blockIdx.x;
  const int b = bh / Hq, h = bh % Hq;

  // init state H0 (input layout (b,h,d,e)) into Hs[e][d]
  {
    const float* h0 = hidden0 + (size_t)bh * (Dq * Dq);
    for (int idx = tid; idx < Dq * Dq; idx += 128) {
      int d = idx >> 6, e = idx & 63;
      Hs[e][d] = h0[idx];
    }
  }
  __syncthreads();

  for (int c = 0; c < NCHUNK; ++c) {
    const int t0 = c * 64;

    // ---- stage gates + inclusive prefix-scan of log f ----
    if (tid < 64) {
      const float* g = gates + ((size_t)(b * Sq + t0 + tid)) * 48;
      ii[tid] = g[h];
      la[tid] = __logf(g[Hq + h]);
      oo[tid] = g[2 * Hq + h];
    }
    __syncthreads();
    for (int off = 1; off < 64; off <<= 1) {
      float v = 0.f;
      if (tid < 64 && tid >= off) v = la[tid - off];
      __syncthreads();
      if (tid < 64 && tid >= off) la[tid] += v;
      __syncthreads();
    }

    // ---- stage Q~, K, K~^T, V^T ----
    {
      const int row = tid >> 1, half = tid & 1;
      const unsigned short* qp =
          qkv + ((size_t)(b * Sq + t0 + row)) * O3 + h * Dq + half * 32;
      const unsigned short* kp = qp + Hq * Dq;
      const unsigned short* vp = qp + 2 * Hq * Dq;
      const float At = __expf(la[row]);
      const float wk = __expf(la[63] - la[row]) * ii[row];
      for (int j = 0; j < 32; j++) {
        int dc = half * 32 + j;
        unsigned short qu = qp[j], ku = kp[j], vu = vp[j];
        Qs[row][dc] = f2bf(At * bf2f(qu));
        Kr[row][dc] = ku;
        Kt[dc][row] = f2bf(wk * bf2f(ku));
        Vt[dc][row] = vu;
      }
    }
    __syncthreads();

    // ---- GEMM1: S' = Q~ K^T ; mask with e^{-la_s} i_s, s<=t ; -> Pm ----
    {
      BFrag aq[2];
#pragma unroll
      for (int k2 = 0; k2 < 2; k2++)
        load_afrag(aq[k2], &Qs[16 * wave + lm][k2 * 32], hf);
#pragma unroll
      for (int ni = 0; ni < 4; ni++) {
        BFrag bk[2];
#pragma unroll
        for (int k2 = 0; k2 < 2; k2++)
          load_bfrag(bk[k2], &Kr[16 * ni + lm][k2 * 32], hf);
        f32x8 acc = {0.f, 0.f, 0.f, 0.f, 0.f, 0.f, 0.f, 0.f};
        acc = wmma_bf16(aq[0], bk[0], acc);
        acc = wmma_bf16(aq[1], bk[1], acc);
        const int sc = 16 * ni + lm;
        const float cs = __expf(-la[sc]) * ii[sc];
#pragma unroll
        for (int r = 0; r < 8; r++) {
          int tr = 16 * wave + 8 * hf + r;
          float f = (sc <= tr) ? cs : 0.f;
          Pm[tr][sc] = f2bf(acc[r] * f);
        }
      }
    }
    __syncthreads();

    // ---- B fragments for this wave's e-columns (V and H0->bf16) ----
    const int ec = 16 * wave + lm;
    BFrag bv[2], bh_[2];
#pragma unroll
    for (int k2 = 0; k2 < 2; k2++) {
      load_bfrag(bv[k2], &Vt[ec][k2 * 32], hf);
#pragma unroll
      for (int i2 = 0; i2 < 16; i2++)
        bh_[k2].u[i2] = f2bf(Hs[ec][k2 * 32 + 16 * hf + i2]);
    }

    // ---- GEMM2+3: O = P.V + Q~.H0 ; scale by o_t ; store att (bf16) ----
#pragma unroll
    for (int mi = 0; mi < 4; mi++) {
      BFrag ap[2], aq2[2];
#pragma unroll
      for (int k2 = 0; k2 < 2; k2++) {
        load_afrag(ap[k2], &Pm[16 * mi + lm][k2 * 32], hf);
        load_afrag(aq2[k2], &Qs[16 * mi + lm][k2 * 32], hf);
      }
      f32x8 acc = {0.f, 0.f, 0.f, 0.f, 0.f, 0.f, 0.f, 0.f};
      acc = wmma_bf16(ap[0], bv[0], acc);
      acc = wmma_bf16(ap[1], bv[1], acc);
      acc = wmma_bf16(aq2[0], bh_[0], acc);
      acc = wmma_bf16(aq2[1], bh_[1], acc);
#pragma unroll
      for (int r = 0; r < 8; r++) {
        int tr = 16 * mi + 8 * hf + r;
        att[((size_t)(b * Sq + t0 + tr)) * (Hq * Dq) + h * Dq + ec] =
            f2bf(oo[tr] * acc[r]);
      }
    }

    // ---- GEMM-H: H' = e^{la63} H0 + K~^T V (wave owns its e-columns) ----
    {
      const float decay = __expf(la[63]);
#pragma unroll
      for (int mi = 0; mi < 4; mi++) {
        BFrag ak[2];
#pragma unroll
        for (int k2 = 0; k2 < 2; k2++)
          load_afrag(ak[k2], &Kt[16 * mi + lm][k2 * 32], hf);
        f32x8 acc;
#pragma unroll
        for (int r = 0; r < 8; r++)
          acc[r] = decay * Hs[ec][16 * mi + 8 * hf + r];
        acc = wmma_bf16(ak[0], bv[0], acc);
        acc = wmma_bf16(ak[1], bv[1], acc);
#pragma unroll
        for (int r = 0; r < 8; r++) Hs[ec][16 * mi + 8 * hf + r] = acc[r];
      }
    }
    __syncthreads();
  }

  // ---- write h_final (b,h,d,e) ----
  {
    float* ho = hfinal + (size_t)bh * (Dq * Dq);
    for (int idx = tid; idx < Dq * Dq; idx += 128) {
      int d = idx >> 6, e = idx & 63;
      ho[idx] = Hs[e][d];
    }
  }
}

// =====================================================================
// Host-side orchestration
// =====================================================================
extern "C" void kernel_launch(void* const* d_in, const int* in_sizes, int n_in,
                              void* d_out, int out_size, void* d_ws,
                              size_t ws_size, hipStream_t stream) {
  const float* x      = (const float*)d_in[0]; // (B,S,DM)
  const float* hidden = (const float*)d_in[1]; // (B,H,D,D)
  const float* w_qkv  = (const float*)d_in[2]; // (3072,1024)
  const float* w_gate = (const float*)d_in[3]; // (48,1024)
  const float* w_out  = (const float*)d_in[4]; // (1024,1024)
  const float* ln_w   = (const float*)d_in[5]; // (1024,)

  float* y = (float*)d_out;                              // B*S*DM
  float* hfinal = y + (size_t)Bq * Sq * DMq;             // B*H*D*D

  char* ws = (char*)d_ws;
  size_t off = 0;
  unsigned short* xn_bf = (unsigned short*)(ws + off);   off += (size_t)Bq * Sq * DMq * 2;
  unsigned short* wqkv_bf = (unsigned short*)(ws + off); off += (size_t)O3 * DMq * 2;
  unsigned short* wout_bf = (unsigned short*)(ws + off); off += (size_t)DMq * DMq * 2;
  unsigned short* qkv_bf = (unsigned short*)(ws + off);  off += (size_t)Bq * Sq * O3 * 2;
  float* gates = (float*)(ws + off);                     off += (size_t)Bq * Sq * 48 * 4;
  unsigned short* att_bf = (unsigned short*)(ws + off);  off += (size_t)Bq * Sq * DMq * 2;
  (void)ws_size; (void)in_sizes; (void)n_in; (void)out_size;

  // 1) RMSNorm + gates
  rmsnorm_gates_kernel<<<Bq * Sq, 256, 0, stream>>>(x, ln_w, w_gate, xn_bf, gates);

  // 2) weight conversions
  {
    int n1 = O3 * DMq;
    cvt_bf16_kernel<<<(n1 + 255) / 256, 256, 0, stream>>>(w_qkv, wqkv_bf, n1);
    int n2 = DMq * DMq;
    cvt_bf16_kernel<<<(n2 + 255) / 256, 256, 0, stream>>>(w_out, wout_bf, n2);
  }

  // 3) QKV projection: (8192 x 1024) x (3072 x 1024)^T -> bf16
  gemm_bf16_nt<0><<<dim3((Bq * Sq) / 128, O3 / 128), 256, 0, stream>>>(
      xn_bf, wqkv_bf, qkv_bf, nullptr, nullptr, O3);

  // 4) chunked mLSTM scan
  mlstm_scan_kernel<<<Bq * Hq, 128, 0, stream>>>(qkv_bf, gates, hidden, att_bf,
                                                 hfinal);

  // 5) output projection + residual: y = att @ w_out^T + x
  gemm_bf16_nt<1><<<dim3((Bq * Sq) / 128, DMq / 128), 256, 0, stream>>>(
      att_bf, wout_bf, nullptr, y, x, DMq);
}
